// Self_Attention_67473936220248
// MI455X (gfx1250) — matrix operand order repaired and checked by reading, not compiled, not run on hardware
//
#include <hip/hip_runtime.h>
#include <hip/hip_bf16.h>

typedef __bf16 bf16_t;
typedef __attribute__((ext_vector_type(16))) __bf16 v16bf;
typedef __attribute__((ext_vector_type(8)))  __bf16 v8bf;
typedef __attribute__((ext_vector_type(8)))  float  v8f;

#define B_ 8
#define N_ 4096
#define C_ 1024
#define H_ 16
#define D_ 64
#define SCALE_ 0.125f   // 64^-0.5

__device__ __forceinline__ v8f wmma_bf16(v16bf a, v16bf b, v8f c) {
  return __builtin_amdgcn_wmma_f32_16x16x32_bf16(false, a, false, b, (short)0, c,
                                                 false, false);
}

// Build a 16-element bf16 fragment from two 8-element (16B) chunks.
__device__ __forceinline__ v16bf frag16(const bf16_t* p0, const bf16_t* p1) {
  v8bf lo = *(const v8bf*)p0;
  v8bf hi = *(const v8bf*)p1;
  v16bf r;
#pragma unroll
  for (int i = 0; i < 8; ++i) { r[i] = lo[i]; r[i + 8] = hi[i]; }
  return r;
}

// CDNA5 async copy: 16 bytes global -> LDS, tracked by ASYNCcnt.
__device__ __forceinline__ void async_copy16(const bf16_t* g, bf16_t* l) {
  unsigned loff = (unsigned)(unsigned long long)l;   // AS3 offset in low 32 bits
  unsigned long long ga = (unsigned long long)g;
  asm volatile("global_load_async_to_lds_b128 %0, %1, off"
               :: "v"(loff), "v"(ga)
               : "memory");
}

__global__ __launch_bounds__(256) void cvt_bf16(const float* __restrict__ s,
                                                bf16_t* __restrict__ d, int n) {
  int i = blockIdx.x * blockDim.x + threadIdx.x;
  int stride = gridDim.x * blockDim.x;
  for (; i < n; i += stride) d[i] = (bf16_t)s[i];
}

// y = A(M=32768 x K=1024) @ W^T  (W is 1024 x 1024 row-major).
// 128x128 block tile, BK=32, double-buffered LDS via async-to-LDS copies.
// 8 waves, each wave: 2x4 tiles of 16x16.
// MODE 0: bf16, packed b128 stores to (b,h,d,n)   [q, k]
// MODE 1: bf16 scatter to (b,h,n,e)               [v]
// MODE 2: f32 + bias, row-major (m, o)            [projection -> d_out]
template <int MODE>
__global__ __launch_bounds__(256) void gemm_bf16(
    const bf16_t* __restrict__ A, const bf16_t* __restrict__ W,
    void* __restrict__ dst, const float* __restrict__ bias) {
  __shared__ bf16_t As[2][128][48];   // 32 + 16 pad, rows stay 16B aligned
  __shared__ bf16_t Ws[2][128][48];

  const int tid  = threadIdx.x;
  const int lane = tid & 31, wid = tid >> 5;
  const int lrow = lane & 15, hs  = lane >> 4;     // half-wave select
  const int mBase = blockIdx.y * 128;
  const int oBase = blockIdx.x * 128;
  const int waveM = (wid & 3) * 32;
  const int waveO = (wid >> 2) * 64;

  const int ldRow = tid >> 2;          // 0..63
  const int ldCol = (tid & 3) * 8;     // 0,8,16,24

  v8f acc[2][4] = {};

  // Prologue: issue tile 0 (4 async b128 ops per thread).
#pragma unroll
  for (int r2 = 0; r2 < 2; ++r2) {
    int row = ldRow + r2 * 64;
    async_copy16(&A[(size_t)(mBase + row) * 1024 + ldCol], &As[0][row][ldCol]);
    async_copy16(&W[(size_t)(oBase + row) * 1024 + ldCol], &Ws[0][row][ldCol]);
  }

  for (int ks = 0; ks < 32; ++ks) {
    const int cur = ks & 1;
    if (ks + 1 < 32) {
      const int k1 = (ks + 1) * 32;
#pragma unroll
      for (int r2 = 0; r2 < 2; ++r2) {
        int row = ldRow + r2 * 64;
        async_copy16(&A[(size_t)(mBase + row) * 1024 + k1 + ldCol],
                     &As[cur ^ 1][row][ldCol]);
        async_copy16(&W[(size_t)(oBase + row) * 1024 + k1 + ldCol],
                     &Ws[cur ^ 1][row][ldCol]);
      }
      asm volatile("s_wait_asynccnt 0x4" ::: "memory");  // oldest tile landed
    } else {
      asm volatile("s_wait_asynccnt 0x0" ::: "memory");
    }
    __syncthreads();

    v16bf af[2], wf[4];
#pragma unroll
    for (int mi = 0; mi < 2; ++mi) {
      const bf16_t* p = &As[cur][waveM + mi * 16 + lrow][hs * 8];
      af[mi] = frag16(p, p + 16);          // A: K = hs*8..+7, hs*8+16..+23
    }
#pragma unroll
    for (int ni = 0; ni < 4; ++ni) {
      const bf16_t* p = &Ws[cur][waveO + ni * 16 + lrow][hs * 8];
      wf[ni] = frag16(p, p + 16);          // B: K = hs*8..+7, hs*8+16..+23
    }
#pragma unroll
    for (int mi = 0; mi < 2; ++mi)
#pragma unroll
      for (int ni = 0; ni < 4; ++ni)
        acc[mi][ni] = wmma_bf16(af[mi], wf[ni], acc[mi][ni]);
    __syncthreads();   // all readers done before this buffer is re-filled
  }

  if (MODE == 0) {
    // Pack the 8 accumulator rows (consecutive n) into one b128 store.
#pragma unroll
    for (int mi = 0; mi < 2; ++mi)
#pragma unroll
      for (int ni = 0; ni < 4; ++ni) {
        v8bf pk;
#pragma unroll
        for (int r = 0; r < 8; ++r) pk[r] = (bf16_t)acc[mi][ni][r];
        int m0 = mBase + waveM + mi * 16 + hs * 8;  // n varies along r
        int o  = oBase + waveO + ni * 16 + lrow;
        int b = m0 >> 12, n0 = m0 & 4095;
        int h = o >> 6,  e = o & 63;
        *(v8bf*)&((bf16_t*)dst)[((((size_t)(b * 16 + h)) * 64 + e) << 12) + n0] = pk;
      }
  } else {
#pragma unroll
    for (int mi = 0; mi < 2; ++mi)
#pragma unroll
      for (int ni = 0; ni < 4; ++ni)
#pragma unroll
        for (int r = 0; r < 8; ++r) {
          int m = mBase + waveM + mi * 16 + r + hs * 8;
          int o = oBase + waveO + ni * 16 + lrow;
          float v = acc[mi][ni][r];
          if (MODE == 2) {
            ((float*)dst)[(size_t)m * 1024 + o] = v + bias[o];
          } else {  // MODE 1: v in (b,h,n,e)
            int b = m >> 12, n = m & 4095;
            int h = o >> 6,  e = o & 63;
            ((bf16_t*)dst)[((((size_t)(b * 16 + h)) << 12) + n) * 64 + e] =
                (bf16_t)v;
          }
        }
  }
}

// One workgroup per (b,h). S = scale * Q(64xN) Kt(Nx64); softmax rows;
// O = P(64x64) @ V(64xN); store O into (b, d, h, n) flat layout.
__global__ __launch_bounds__(256) void attn_kernel(
    const bf16_t* __restrict__ qT, const bf16_t* __restrict__ kT,
    const bf16_t* __restrict__ v2, bf16_t* __restrict__ aout) {
  __shared__ float  Sf[64][68];
  __shared__ bf16_t Pb[64][80];   // 160B rows: 16B aligned for b128 reads

  const int bh = blockIdx.x;
  const int b = bh >> 4, h = bh & 15;
  const int tid  = threadIdx.x;
  const int lane = tid & 31, wid = tid >> 5;
  const int lrow = lane & 15, hs  = lane >> 4;

  const bf16_t* Q  = qT + (size_t)bh * 64 * 4096;
  const bf16_t* Kp = kT + (size_t)bh * 64 * 4096;
  const bf16_t* Vp = v2 + (size_t)bh * 4096 * 64;

  // ---- Phase A: S tiles, 2 per wave, K-reduction over N=4096 ----
  const int dT = wid >> 1;
  const int eT = (wid & 1) * 2;
  v8f s0 = {}, s1 = {};
  for (int n0 = 0; n0 < 4096; n0 += 32) {
    const bf16_t* ap = Q + (size_t)(dT * 16 + lrow) * 4096 + n0 + hs * 8;
    v16bf a = frag16(ap, ap + 16);
    const bf16_t* b0 = Kp + (size_t)(eT * 16 + lrow) * 4096 + n0 + hs * 8;
    const bf16_t* b1 = b0 + (size_t)16 * 4096;
    v16bf f0 = frag16(b0, b0 + 16);
    v16bf f1 = frag16(b1, b1 + 16);
    s0 = wmma_bf16(a, f0, s0);
    s1 = wmma_bf16(a, f1, s1);
  }
#pragma unroll
  for (int r = 0; r < 8; ++r) {
    int d = dT * 16 + r + hs * 8;
    int e = eT * 16 + lrow;
    Sf[d][e]      = s0[r] * SCALE_;
    Sf[d][e + 16] = s1[r] * SCALE_;
  }
  __syncthreads();

  // ---- Phase B: softmax over 64-wide rows (one thread per row) ----
  if (tid < 64) {
    float mx = -1e30f;
#pragma unroll
    for (int e = 0; e < 64; ++e) mx = fmaxf(mx, Sf[tid][e]);
    float sum = 0.f;
#pragma unroll
    for (int e = 0; e < 64; ++e) sum += __expf(Sf[tid][e] - mx);
    float inv = 1.f / sum;
#pragma unroll
    for (int e = 0; e < 64; ++e)
      Pb[tid][e] = (bf16_t)(__expf(Sf[tid][e] - mx) * inv);
  }
  __syncthreads();

  // ---- Phase C: O = P @ V, each wave owns 512 columns of n ----
  v16bf pf[4][2];
#pragma unroll
  for (int dt = 0; dt < 4; ++dt)
#pragma unroll
    for (int ks = 0; ks < 2; ++ks) {
      const bf16_t* p = &Pb[dt * 16 + lrow][ks * 32 + hs * 8];
      pf[dt][ks] = frag16(p, p + 16);
    }
  const int nBase = wid * 512;
  for (int nt = 0; nt < 32; ++nt) {
    int ncol = nBase + nt * 16 + lrow;
    v8f o[4] = {};
#pragma unroll
    for (int ks = 0; ks < 2; ++ks) {
      const bf16_t* vp = Vp + (size_t)ncol * 64 + ks * 32 + hs * 8;
      v16bf vf = frag16(vp, vp + 16);
#pragma unroll
      for (int dt = 0; dt < 4; ++dt) o[dt] = wmma_bf16(pf[dt][ks], vf, o[dt]);
    }
#pragma unroll
    for (int dt = 0; dt < 4; ++dt)
#pragma unroll
      for (int r = 0; r < 8; ++r) {
        int d = dt * 16 + r + hs * 8;
        aout[(((size_t)(b * 64 + d) * 16 + h) << 12) + ncol] = (bf16_t)o[dt][r];
      }
  }
}

extern "C" void kernel_launch(void* const* d_in, const int* in_sizes, int n_in,
                              void* d_out, int out_size, void* d_ws, size_t ws_size,
                              hipStream_t stream) {
  const float* x  = (const float*)d_in[0];
  const float* Wq = (const float*)d_in[1];
  const float* Wk = (const float*)d_in[2];
  const float* Wv = (const float*)d_in[3];
  const float* Wp = (const float*)d_in[4];
  const float* bp = (const float*)d_in[5];

  const size_t XE = (size_t)B_ * N_ * C_;  // 33,554,432 elems
  const size_t WE = (size_t)C_ * C_;       //  1,048,576 elems

  bf16_t* xb  = (bf16_t*)d_ws;
  bf16_t* wqb = xb  + XE;
  bf16_t* wkb = wqb + WE;
  bf16_t* wvb = wkb + WE;
  bf16_t* wpb = wvb + WE;
  bf16_t* qT  = wpb + WE;
  bf16_t* kT  = qT  + XE;
  bf16_t* v2  = kT  + XE;
  bf16_t* aout = xb;   // xb is dead after the q/k/v GEMMs; reuse for attn out

  cvt_bf16<<<4096, 256, 0, stream>>>(x,  xb,  (int)XE);
  cvt_bf16<<<512,  256, 0, stream>>>(Wq, wqb, (int)WE);
  cvt_bf16<<<512,  256, 0, stream>>>(Wk, wkb, (int)WE);
  cvt_bf16<<<512,  256, 0, stream>>>(Wv, wvb, (int)WE);
  cvt_bf16<<<512,  256, 0, stream>>>(Wp, wpb, (int)WE);

  dim3 gg(8, 256);   // O/128, M/128
  gemm_bf16<0><<<gg, 256, 0, stream>>>(xb, wqb, qT, nullptr);
  gemm_bf16<0><<<gg, 256, 0, stream>>>(xb, wkb, kT, nullptr);
  gemm_bf16<1><<<gg, 256, 0, stream>>>(xb, wvb, v2, nullptr);

  attn_kernel<<<128, 256, 0, stream>>>(qT, kT, v2, aout);

  gemm_bf16<2><<<gg, 256, 0, stream>>>(aout, wpb, d_out, bp);
}
